// ROI_Detector_49177375539819
// MI455X (gfx1250) — matrix operand order repaired and checked by reading, not compiled, not run on hardware
//
#include <hip/hip_runtime.h>
#include <hip/hip_bf16.h>
#include <math.h>

// ---------------- types ----------------
typedef _Float16 v16h __attribute__((ext_vector_type(16)));
typedef _Float16 v8h  __attribute__((ext_vector_type(8)));
typedef float    v8f  __attribute__((ext_vector_type(8)));

#define NPROP 1024
#define CCH   512
#define HF    50
#define WF    50
#define PP    7
#define FC    2048
#define KIN   (CCH * PP * PP)   // 25088
#define NCLS  21
#define NBOX  80
#define NEGC  (-1e30f)

static __device__ __forceinline__ v16h cat8(v8h lo, v8h hi) {
    return __builtin_shufflevector(lo, hi, 0,1,2,3,4,5,6,7,8,9,10,11,12,13,14,15);
}
// A fragment (16x32 f16, ISA 7.12.2): per-lane runs [0..7] and [16..23] from base
static __device__ __forceinline__ v16h ldA(const _Float16* p) {
    return cat8(*(const v8h*)p, *(const v8h*)(p + 16));
}
// B fragment (32x16 f16): per-lane runs [0..7] and [16..23] from base (same k order as A)
static __device__ __forceinline__ v16h ldB(const _Float16* p) {
    return cat8(*(const v8h*)p, *(const v8h*)(p + 16));
}
static __device__ __forceinline__ v8f wmma(v16h a, v16h b, v8f c) {
    return __builtin_amdgcn_wmma_f32_16x16x32_f16(false, a, false, b, (short)0, c, false, false);
}

// ---------------- f32 -> f16 conversion (one HBM pass; NT loads keep L2 for the f16 copy) -
__global__ __launch_bounds__(256) void cvt_f32_to_f16(const float* __restrict__ src,
                                                      _Float16* __restrict__ dst, int n) {
    int stride = gridDim.x * 256;
    for (int i = blockIdx.x * 256 + threadIdx.x; i < n; i += stride) {
        float v = __builtin_nontemporal_load(src + i);   // TH=NT: don't evict f16 working set
        dst[i] = (_Float16)v;
    }
}

// ---------------- ROI max-pool (torchvision roi_pool semantics), emits f16 features ------
__global__ __launch_bounds__(256) void roi_pool_kernel(const float* __restrict__ feat,
                                                       const float* __restrict__ props,
                                                       const int* __restrict__ image_h_p,
                                                       _Float16* __restrict__ out) {
    __shared__ int sws[49], swe[49], shs[49], she[49];
    const int n = blockIdx.x;
    const float img_h = (float)image_h_p[0];
    const float scale = exp2f(rintf(log2f((float)HF / img_h)));   // 1/16 here

    const int x1 = (int)rintf(props[n * 4 + 0] * scale);
    const int y1 = (int)rintf(props[n * 4 + 1] * scale);
    const int x2 = (int)rintf(props[n * 4 + 2] * scale);
    const int y2 = (int)rintf(props[n * 4 + 3] * scale);
    const float roi_w = (float)max(x2 - x1 + 1, 1);
    const float roi_h = (float)max(y2 - y1 + 1, 1);
    const float bw = roi_w / (float)PP;
    const float bh = roi_h / (float)PP;

    const int t = threadIdx.x;
    if (t < 49) {
        int ph = t / PP, pw = t % PP;
        sws[t] = min(max(x1 + (int)floorf((float)pw * bw), 0), WF);
        swe[t] = min(max(x1 + (int)ceilf((float)(pw + 1) * bw), 0), WF);
        shs[t] = min(max(y1 + (int)floorf((float)ph * bh), 0), HF);
        she[t] = min(max(y1 + (int)ceilf((float)(ph + 1) * bh), 0), HF);
    }
    __syncthreads();

    for (int i = t; i < CCH * 49; i += 256) {
        int c = i / 49, bin = i % 49;
        int ws = sws[bin], we = swe[bin], hs = shs[bin], he = she[bin];
        const float* fp = feat + (size_t)c * (HF * WF);
        float m = NEGC;
        for (int h = hs; h < he; ++h)
            for (int w = ws; w < we; ++w)
                m = fmaxf(m, fp[h * WF + w]);
        if (m < NEGC * 0.5f) m = 0.0f;   // empty-bin -> 0 per reference
        out[(size_t)n * KIN + i] = (_Float16)m;
    }
}

// ---------------- WMMA GEMM: C[M,N] = A[M,K] * B[N,K]^T (+bias, opt ReLU) ----------------
// Wave tile 32M x 64N (2 A-frags x 4 B-frags -> 8 v_wmma per 32-K step), software-pipelined
// ping-pong over 64-K so s_wait_loadcnt covers loads issued a full stage earlier.
// Block = 8 waves (2 Mrows x 4 Ncols) -> 64M x 256N. K must be a multiple of 64.
__global__ __launch_bounds__(256) void gemm_wmma_f16(const _Float16* __restrict__ A,
                                                     const _Float16* __restrict__ B,
                                                     const float* __restrict__ bias,
                                                     void* __restrict__ Cout,
                                                     int M, int N, int K, int relu_f16out) {
    const int lane = threadIdx.x & 31;
    const int wave = threadIdx.x >> 5;
    const int m0    = blockIdx.y * 64 + (wave >> 2) * 32;
    const int nbase = blockIdx.x * 256 + (wave & 3) * 64;
    const int half  = lane >> 4;     // 0/1 lane group
    const int l15   = lane & 15;

    const _Float16* Ap0 = A + (size_t)(m0 + l15) * K + half * 8;
    const _Float16* Ap1 = Ap0 + (size_t)16 * K;

    const _Float16* Bp[4];
    int ncol[4];
#pragma unroll
    for (int tI = 0; tI < 4; ++tI) {
        int n = nbase + tI * 16 + l15;
        ncol[tI] = n;
        int nc = n < N ? n : (N - 1);            // clamp for safe loads; store is guarded
        Bp[tI] = B + (size_t)nc * K + half * 8;
    }

    v8f acc[2][4];
#pragma unroll
    for (int f = 0; f < 2; ++f)
#pragma unroll
        for (int tI = 0; tI < 4; ++tI) acc[f][tI] = v8f{};

    // ---- prologue: stage-0 fragments ----
    v16h A0 = ldA(Ap0), A1 = ldA(Ap1);
    v16h B0 = ldB(Bp[0]), B1 = ldB(Bp[1]), B2 = ldB(Bp[2]), B3 = ldB(Bp[3]);

    for (int k0 = 0; k0 < K; k0 += 64) {
        __builtin_prefetch(Bp[0] + k0 + 2048, 0, 1);   // global_prefetch_b8 far ahead
        __builtin_prefetch(Bp[2] + k0 + 2048, 0, 1);

        // issue stage-1 loads (k0+32) before consuming stage-0 fragments
        v16h A0n = ldA(Ap0 + k0 + 32), A1n = ldA(Ap1 + k0 + 32);
        v16h B0n = ldB(Bp[0] + k0 + 32), B1n = ldB(Bp[1] + k0 + 32);
        v16h B2n = ldB(Bp[2] + k0 + 32), B3n = ldB(Bp[3] + k0 + 32);

        acc[0][0] = wmma(A0, B0, acc[0][0]);
        acc[0][1] = wmma(A0, B1, acc[0][1]);
        acc[0][2] = wmma(A0, B2, acc[0][2]);
        acc[0][3] = wmma(A0, B3, acc[0][3]);
        acc[1][0] = wmma(A1, B0, acc[1][0]);
        acc[1][1] = wmma(A1, B1, acc[1][1]);
        acc[1][2] = wmma(A1, B2, acc[1][2]);
        acc[1][3] = wmma(A1, B3, acc[1][3]);

        // issue next iteration's stage-0 loads (k0+64), uniform guard keeps EXEC full
        if (k0 + 64 < K) {
            A0 = ldA(Ap0 + k0 + 64); A1 = ldA(Ap1 + k0 + 64);
            B0 = ldB(Bp[0] + k0 + 64); B1 = ldB(Bp[1] + k0 + 64);
            B2 = ldB(Bp[2] + k0 + 64); B3 = ldB(Bp[3] + k0 + 64);
        }

        acc[0][0] = wmma(A0n, B0n, acc[0][0]);
        acc[0][1] = wmma(A0n, B1n, acc[0][1]);
        acc[0][2] = wmma(A0n, B2n, acc[0][2]);
        acc[0][3] = wmma(A0n, B3n, acc[0][3]);
        acc[1][0] = wmma(A1n, B0n, acc[1][0]);
        acc[1][1] = wmma(A1n, B1n, acc[1][1]);
        acc[1][2] = wmma(A1n, B2n, acc[1][2]);
        acc[1][3] = wmma(A1n, B3n, acc[1][3]);
    }

    // C layout: VGPR r -> row = m0 + f*16 + half*8 + r, col = lane&15 within tile
#pragma unroll
    for (int tI = 0; tI < 4; ++tI) {
        int col = ncol[tI];
        if (col < N) {
            float bv = bias[col];
#pragma unroll
            for (int f = 0; f < 2; ++f) {
#pragma unroll
                for (int r = 0; r < 8; ++r) {
                    int row = m0 + f * 16 + half * 8 + r;
                    float v = acc[f][tI][r] + bv;
                    if (relu_f16out) {
                        v = v > 0.0f ? v : 0.0f;
                        ((_Float16*)Cout)[(size_t)row * N + col] = (_Float16)v;
                    } else {
                        ((float*)Cout)[(size_t)row * N + col] = v;
                    }
                }
            }
        }
    }
}

// ---------------- post-processing + greedy class-wise NMS (one 1024-thread block) --------
__global__ __launch_bounds__(1024) void postproc_nms_kernel(const float* __restrict__ cls,
                                                            const float* __restrict__ deltas,
                                                            const float* __restrict__ props,
                                                            const int* __restrict__ ih_p,
                                                            const int* __restrict__ iw_p,
                                                            float* __restrict__ out) {
    __shared__ float nx1[NPROP], ny1[NPROP], nx2[NPROP], ny2[NPROP];
    __shared__ float ssc[NPROP];
    __shared__ int   ssup[NPROP];
    __shared__ float rv[32];
    __shared__ int   ri[32];

    const int t = threadIdx.x;
    const float ih = (float)ih_p[0];
    const float iw = (float)iw_p[0];

    // softmax / argmax (argmax(probs) == argmax(scores); score = 1/sum(exp(s - smax)))
    const float* sc = cls + (size_t)t * NCLS;
    float smax = sc[0]; int lab = 0;
    for (int j = 1; j < NCLS; ++j)
        if (sc[j] > smax) { smax = sc[j]; lab = j; }
    float sum = 0.0f;
    for (int j = 0; j < NCLS; ++j) sum += expf(sc[j] - smax);
    float score = 1.0f / sum;

    int ci = lab - 1; if (ci < 0) ci = 0;
    const float* d = deltas + (size_t)t * NBOX + ci * 4;
    float p0 = props[t * 4 + 0], p1 = props[t * 4 + 1];
    float p2 = props[t * 4 + 2], p3 = props[t * 4 + 3];
    float pw = p2 - p0, ph = p3 - p1;
    float cx = p0 + 0.5f * pw, cy = p1 + 0.5f * ph;
    float gcx = d[0] * pw + cx, gcy = d[1] * ph + cy;
    float gw = expf(d[2]) * pw, gh = expf(d[3]) * ph;
    float bx1 = fminf(fmaxf(gcx - 0.5f * gw, 0.0f), iw);
    float by1 = fminf(fmaxf(gcy - 0.5f * gh, 0.0f), ih);
    float bx2 = fminf(fmaxf(gcx + 0.5f * gw, 0.0f), iw);
    float by2 = fminf(fmaxf(gcy + 0.5f * gh, 0.0f), ih);

    bool valid = (lab > 0) && (score > 0.05f);
    float off = (float)lab * (fmaxf(ih, iw) + 1.0f);
    float mx1 = bx1 + off, my1 = by1 + off, mx2 = bx2 + off, my2 = by2 + off;
    float marea = fmaxf(mx2 - mx1, 0.0f) * fmaxf(my2 - my1, 0.0f);

    nx1[t] = mx1; ny1[t] = my1; nx2[t] = mx2; ny2[t] = my2;
    ssc[t] = score;
    ssup[t] = valid ? 0 : 1;
    bool keep = false;
    __syncthreads();

    for (int it = 0; it < NPROP; ++it) {
        __syncthreads();
        float v = ssup[t] ? -1.0f : ssc[t];
        int idx = t;
        // intra-wave argmax (wave32), tie-break to lowest index (matches jnp.argmax)
        for (int o = 16; o > 0; o >>= 1) {
            float ov = __shfl_xor(v, o, 32);
            int   oi = __shfl_xor(idx, o, 32);
            if (ov > v || (ov == v && oi < idx)) { v = ov; idx = oi; }
        }
        if ((t & 31) == 0) { rv[t >> 5] = v; ri[t >> 5] = idx; }
        __syncthreads();
        if (t < 32) {
            float v2 = rv[t]; int i2 = ri[t];
            for (int o = 16; o > 0; o >>= 1) {
                float ov = __shfl_xor(v2, o, 32);
                int   oi = __shfl_xor(i2, o, 32);
                if (ov > v2 || (ov == v2 && oi < i2)) { v2 = ov; i2 = oi; }
            }
            if (t == 0) { rv[0] = v2; ri[0] = i2; }
        }
        __syncthreads();
        float best = rv[0];
        int bidx = ri[0];
        if (best > 0.0f) {
            float b0 = nx1[bidx], b1 = ny1[bidx], b2 = nx2[bidx], b3 = ny2[bidx];
            float xx1 = fmaxf(b0, mx1), yy1 = fmaxf(b1, my1);
            float xx2 = fminf(b2, mx2), yy2 = fminf(b3, my2);
            float inter = fmaxf(xx2 - xx1, 0.0f) * fmaxf(yy2 - yy1, 0.0f);
            float a1 = fmaxf(b2 - b0, 0.0f) * fmaxf(b3 - b1, 0.0f);
            float iou = inter / fmaxf(a1 + marea - inter, 1e-6f);
            if (iou > 0.5f) ssup[t] = 1;       // includes self (iou==1)
            if (t == bidx) keep = true;
        }
    }

    // outputs concatenated flat: box[1024*4], score[1024], labels[1024], keep[1024]
    out[t * 4 + 0] = bx1;
    out[t * 4 + 1] = by1;
    out[t * 4 + 2] = bx2;
    out[t * 4 + 3] = by2;
    out[4 * NPROP + t] = score;
    out[5 * NPROP + t] = (float)lab;
    out[6 * NPROP + t] = keep ? 1.0f : 0.0f;
}

// ---------------- launcher ----------------
extern "C" void kernel_launch(void* const* d_in, const int* in_sizes, int n_in,
                              void* d_out, int out_size, void* d_ws, size_t ws_size,
                              hipStream_t stream) {
    const float* feat  = (const float*)d_in[0];
    const float* props = (const float*)d_in[1];
    const float* w6    = (const float*)d_in[2];
    const float* b6    = (const float*)d_in[3];
    const float* w7    = (const float*)d_in[4];
    const float* b7    = (const float*)d_in[5];
    const float* wcls  = (const float*)d_in[6];
    const float* bcls  = (const float*)d_in[7];
    const float* wbox  = (const float*)d_in[8];
    const float* bbox  = (const float*)d_in[9];
    const int*   ih    = (const int*)d_in[10];
    const int*   iwd   = (const int*)d_in[11];

    char* ws = (char*)d_ws;
    size_t off = 0;
    auto alloc = [&](size_t bytes) { char* p = ws + off; off += (bytes + 255) & ~(size_t)255; return p; };
    _Float16* w6h  = (_Float16*)alloc((size_t)FC * KIN * 2);     // 102.8 MB -> L2-resident
    _Float16* w7h  = (_Float16*)alloc((size_t)FC * FC * 2);
    _Float16* wch  = (_Float16*)alloc((size_t)NCLS * FC * 2);
    _Float16* wbh  = (_Float16*)alloc((size_t)NBOX * FC * 2);
    _Float16* roih = (_Float16*)alloc((size_t)NPROP * KIN * 2);  // 51.4 MB, also L2-resident
    _Float16* h6h  = (_Float16*)alloc((size_t)NPROP * FC * 2);
    _Float16* h7h  = (_Float16*)alloc((size_t)NPROP * FC * 2);
    float*    clsS = (float*)alloc((size_t)NPROP * NCLS * 4);
    float*    delS = (float*)alloc((size_t)NPROP * NBOX * 4);
    (void)ws_size; (void)n_in; (void)in_sizes; (void)out_size;

    // 1) weight conversion (single HBM pass over 205MB of w6, NT loads)
    cvt_f32_to_f16<<<2048, 256, 0, stream>>>(w6,   w6h, FC * KIN);
    cvt_f32_to_f16<<<1024, 256, 0, stream>>>(w7,   w7h, FC * FC);
    cvt_f32_to_f16<<<64,   256, 0, stream>>>(wcls, wch, NCLS * FC);
    cvt_f32_to_f16<<<128,  256, 0, stream>>>(wbox, wbh, NBOX * FC);

    // 2) ROI max-pool -> f16 features
    roi_pool_kernel<<<NPROP, 256, 0, stream>>>(feat, props, ih, roih);

    // 3) FC6: [1024,25088] x [2048,25088]^T + b6, ReLU -> f16
    gemm_wmma_f16<<<dim3(FC / 256, NPROP / 64), 256, 0, stream>>>(
        roih, w6h, b6, h6h, NPROP, FC, KIN, 1);
    // 4) FC7
    gemm_wmma_f16<<<dim3(FC / 256, NPROP / 64), 256, 0, stream>>>(
        h6h, w7h, b7, h7h, NPROP, FC, FC, 1);
    // 5) cls head -> f32 scores [1024,21]
    gemm_wmma_f16<<<dim3(1, NPROP / 64), 256, 0, stream>>>(
        h7h, wch, bcls, clsS, NPROP, NCLS, FC, 0);
    // 6) box head -> f32 deltas [1024,80]
    gemm_wmma_f16<<<dim3(1, NPROP / 64), 256, 0, stream>>>(
        h7h, wbh, bbox, delS, NPROP, NBOX, FC, 0);

    // 7) decode + softmax/argmax + NMS
    postproc_nms_kernel<<<1, NPROP, 0, stream>>>(clsS, delS, props, ih, iwd, (float*)d_out);
}
